// Align_fea_80951543595335
// MI455X (gfx1250) — matrix operand order repaired and checked by reading, not compiled, not run on hardware
//
#include <hip/hip_runtime.h>

typedef _Float16 v16h __attribute__((ext_vector_type(16)));
typedef _Float16 v8h  __attribute__((ext_vector_type(8)));
typedef float    v8f  __attribute__((ext_vector_type(8)));
typedef unsigned int u32x4 __attribute__((ext_vector_type(4)));
typedef int          i32x4 __attribute__((ext_vector_type(4)));
typedef int          i32x8 __attribute__((ext_vector_type(8)));

#define B_   4
#define NF_  64
#define H_   128
#define W_   128
#define HW_  (H_ * W_)
#define NSP  (B_ * HW_)       // 65536 spatial positions
#define DG_  8
#define KK_  9

#if defined(__has_builtin)
#if __has_builtin(__builtin_amdgcn_tensor_load_to_lds) && __has_builtin(__builtin_amdgcn_s_wait_tensorcnt)
#define HAVE_TDM 1
#endif
#endif

__device__ __forceinline__ float lrelu_f(float v) { return v >= 0.f ? v : 0.1f * v; }

#ifdef HAVE_TDM
// --------------------------------------------------------------------------
// TDM: DMA a [tile_m rows x tile_k elems] f16 tile (row pitch = pitch elems)
// from global to LDS. D# built per CDNA5 ISA 8.3/8.4 bitfields.
// Must be called by exactly one wave; EXEC is ignored by TENSOR_LOAD_TO_LDS.
// --------------------------------------------------------------------------
__device__ __forceinline__ void tdm_load_2d_f16(_Float16* lds_dst,
                                                const _Float16* gsrc,
                                                unsigned pitch_elems,
                                                unsigned rows,
                                                unsigned tile_k,
                                                unsigned tile_m)
{
    // LDS flat pointers keep the LDS byte address in the low 32 bits.
    unsigned lds_addr = (unsigned)(unsigned long long)lds_dst;
    unsigned long long ga = (unsigned long long)gsrc;

    u32x4 g0;
    g0.x = 1u;                                   // count=1, is_restore=0, no gather
    g0.y = lds_addr;                             // lds_addr [63:32]
    g0.z = (unsigned)ga;                         // global_addr [95:64]
    g0.w = (unsigned)(ga >> 32) | (2u << 30);    // global_addr[56:32] | type=2

    i32x8 g1;
    g1[0] = 0x00010000;                          // workgroup_mask=0, data_size=1 (2B)
    g1[1] = (int)((pitch_elems & 0xFFFFu) << 16);                 // tensor_dim0 lo16
    g1[2] = (int)((pitch_elems >> 16) | ((rows & 0xFFFFu) << 16));// dim0 hi / dim1 lo
    g1[3] = (int)((rows >> 16) | (tile_k << 16));                 // dim1 hi / tile_dim0
    g1[4] = (int)(tile_m & 0xFFFFu);                              // tile_dim1, tile_dim2=0
    g1[5] = (int)pitch_elems;                                     // tensor_dim0_stride lo32
    g1[6] = 0;                                                    // stride0 hi / stride1 lo
    g1[7] = 0;

    i32x4 z4 = {0, 0, 0, 0};
#if __clang_major__ >= 23
    i32x8 z8 = {0, 0, 0, 0, 0, 0, 0, 0};
    __builtin_amdgcn_tensor_load_to_lds(g0, g1, z4, z4, z8, 0);
#else
    __builtin_amdgcn_tensor_load_to_lds(g0, g1, z4, z4, 0);
#endif
}
#endif // HAVE_TDM

// --------------------------------------------------------------------------
// Convert f32 weight matrix [Msrc x K] to f16 [Mdst x K], zero-padding rows.
// --------------------------------------------------------------------------
__global__ __launch_bounds__(256)
void convert_w_f16(const float* __restrict__ src, _Float16* __restrict__ dst,
                   int Msrc, int Mdst, int K)
{
    int idx = blockIdx.x * 256 + threadIdx.x;
    if (idx >= Mdst * K) return;
    int m = idx / K;
    dst[idx] = (_Float16)((m < Msrc) ? src[idx] : 0.f);
}

// --------------------------------------------------------------------------
// Implicit-GEMM 3x3 conv (stride 1, pad 1, NCHW), WMMA f16 -> f32.
// Block: 128 threads = 4 waves. Block tile: 64 (M) x 64 (N).
// Each wave: 16 M rows x 4 N-subtiles (4 accumulators, A fragment reused 4x).
// A (f16 weights, rows padded to 64-multiple) pulled to LDS by TDM.
// --------------------------------------------------------------------------
template<bool RELU>
__global__ __launch_bounds__(128)
void conv3x3_gemm_wmma(const float* __restrict__ in0, const float* __restrict__ in1,
                       int C0, int C,
                       const _Float16* __restrict__ wh, const float* __restrict__ bias,
                       float* __restrict__ out, int Mtot)
{
    const int Ktot = C * 9;
    __shared__ __align__(16) _Float16 As[64][32];   // [M][K]
    __shared__ __align__(16) _Float16 Bs[32][64];   // [K][N]

    const int tid  = threadIdx.x;
    const int lane = tid & 31;
    const int wid  = tid >> 5;
    const int n0   = blockIdx.x * 64;   // 64 consecutive pixels -> same image row
    const int m0   = blockIdx.y * 64;

    v8f acc[4] = {{}, {}, {}, {}};

    for (int kc = 0; kc < Ktot; kc += 32) {
        // ---- A tile: 64x32 f16 weights ----
#ifdef HAVE_TDM
        if (wid == 0) {
            tdm_load_2d_f16(&As[0][0], wh + (size_t)m0 * Ktot + kc,
                            (unsigned)Ktot, 64u, 32u, 64u);
            __builtin_amdgcn_s_wait_tensorcnt(0);
        }
#else
        for (int e = tid; e < 64 * 32; e += 128) {
            int mm = e >> 5, kk = e & 31;
            As[mm][kk] = wh[(size_t)(m0 + mm) * Ktot + kc + kk];
        }
#endif
        // ---- B tile: im2col patch 32(K) x 64(N), 16 elems/thread ----
        #pragma unroll
        for (int r = 0; r < 16; ++r) {
            int e  = tid + r * 128;          // 0..2047
            int kk = e >> 6;
            int nn = e & 63;
            int k  = kc + kk;
            int c  = k / 9, j = k % 9;
            int n  = n0 + nn;
            int b  = n >> 14;                // / HW_
            int rem = n & (HW_ - 1);
            int y  = rem >> 7;               // / W_
            int x  = rem & (W_ - 1);
            int yy = y + (j / 3) - 1;
            int xx = x + (j % 3) - 1;
            float v = 0.f;
            if (yy >= 0 && yy < H_ && xx >= 0 && xx < W_) {
                v = (c < C0) ? in0[((size_t)b * C0 + c) * HW_ + yy * W_ + xx]
                             : in1[((size_t)b * (C - C0) + (c - C0)) * HW_ + yy * W_ + xx];
            }
            Bs[kk][nn] = (_Float16)v;
        }
        __syncthreads();

        // ---- fragments per ISA wave32 layouts ----
        const int mloc = (lane & 15) + wid * 16;
        const int kb   = (lane < 16) ? 0 : 8;
        v16h afrag;
        {
            v8h lo = *(const v8h*)&As[mloc][kb];
            v8h hi = *(const v8h*)&As[mloc][kb + 16];
            #pragma unroll
            for (int i = 0; i < 8; ++i) { afrag[i] = lo[i]; afrag[i + 8] = hi[i]; }
        }
        #pragma unroll
        for (int nt = 0; nt < 4; ++nt) {
            const int ncol = nt * 16 + (lane & 15);
            v16h bfrag;
            #pragma unroll
            for (int i = 0; i < 8; ++i) { bfrag[i] = Bs[kb + i][ncol]; bfrag[i + 8] = Bs[kb + 16 + i][ncol]; }
            acc[nt] = __builtin_amdgcn_wmma_f32_16x16x32_f16(false, afrag, false, bfrag,
                                                             (short)0, acc[nt], false, false);
        }
        __syncthreads();
    }

    // ---- store: VGPR r, lane<16 -> M=r ; lane>=16 -> M=r+8 ; N=lane&15 ----
    const int nn = lane & 15;
    const int mb = m0 + wid * 16 + ((lane < 16) ? 0 : 8);
    #pragma unroll
    for (int nt = 0; nt < 4; ++nt) {
        int n   = n0 + nt * 16 + nn;
        int b   = n >> 14;
        int rem = n & (HW_ - 1);
        #pragma unroll
        for (int r = 0; r < 8; ++r) {
            int m = mb + r;
            if (m < Mtot) {
                float v = acc[nt][r] + bias[m];
                if (RELU) v = lrelu_f(v);
                out[((size_t)b * Mtot + m) * HW_ + rem] = v;
            }
        }
    }
}

// --------------------------------------------------------------------------
// Depthwise 3x3 conv + lrelu, fp32 VALU (memory-bound, stays scalar)
// --------------------------------------------------------------------------
__global__ __launch_bounds__(256)
void dwconv3x3_lrelu(const float* __restrict__ in, const float* __restrict__ w,
                     const float* __restrict__ bias, float* __restrict__ out)
{
    int idx = blockIdx.x * 256 + threadIdx.x;
    if (idx >= B_ * NF_ * HW_) return;
    int x = idx & (W_ - 1);
    int y = (idx >> 7) & (H_ - 1);
    int c = (idx >> 14) & (NF_ - 1);
    const float* plane = in + (size_t)(idx - y * W_ - x);
    float acc = bias[c];
    #pragma unroll
    for (int j = 0; j < 9; ++j) {
        int yy = y + (j / 3) - 1;
        int xx = x + (j % 3) - 1;
        if (yy >= 0 && yy < H_ && xx >= 0 && xx < W_)
            acc += w[c * 9 + j] * plane[yy * W_ + xx];
    }
    out[idx] = lrelu_f(acc);
}

// --------------------------------------------------------------------------
// Modulated deformable 3x3 conv (DCNv2) + lrelu.
// Block tile: 64(M=all out ch) x 64(N pixels). Sampling coords + sigmoid(mask)
// staged once per block; im2col "val" tiles bilinear-sampled to f16 LDS and
// contracted with w_dcn (f16, via TDM) on WMMA.
// om layout: [B, 3*DG*9, H, W]
// --------------------------------------------------------------------------
__global__ __launch_bounds__(128)
void deform_conv_wmma(const float* __restrict__ xin, const float* __restrict__ om,
                      const _Float16* __restrict__ wh, const float* __restrict__ bias,
                      float* __restrict__ out)
{
    __shared__ __align__(16) _Float16 As[64][32];
    __shared__ __align__(16) _Float16 Bs[32][64];
    __shared__ float sPy[DG_ * KK_][64];
    __shared__ float sPx[DG_ * KK_][64];
    __shared__ float sMk[DG_ * KK_][64];

    const int tid  = threadIdx.x;
    const int lane = tid & 31;
    const int wid  = tid >> 5;
    const int n0   = blockIdx.x * 64;
    const int b    = n0 >> 14;
    const int rem0 = n0 & (HW_ - 1);
    const int y    = rem0 >> 7;
    const int x0   = rem0 & (W_ - 1);

    // stage sampling coords + modulation: 72 (dg,kpos) x 64 pixels
    for (int e = tid; e < DG_ * KK_ * 64; e += 128) {
        int i  = e >> 6;            // dg*9 + j
        int nn = e & 63;
        int j  = i % KK_;
        size_t base = (size_t)b * (3 * DG_ * KK_) * HW_ + (size_t)y * W_ + (x0 + nn);
        float oy = om[base + (size_t)i * HW_];
        float ox = om[base + (size_t)(DG_ * KK_ + i) * HW_];
        float ml = om[base + (size_t)(2 * DG_ * KK_ + i) * HW_];
        sPy[i][nn] = (float)y + (float)(j / 3 - 1) + oy;
        sPx[i][nn] = (float)(x0 + nn) + (float)(j % 3 - 1) + ox;
        sMk[i][nn] = 1.f / (1.f + __expf(-ml));
    }
    __syncthreads();

    v8f acc[4] = {{}, {}, {}, {}};
    const int Ktot = NF_ * KK_;     // 576

    for (int kc = 0; kc < Ktot; kc += 32) {
        // ---- A tile: w_dcn rows (flat o*576 + c*9 + j == GEMM k order) ----
#ifdef HAVE_TDM
        if (wid == 0) {
            tdm_load_2d_f16(&As[0][0], wh + (size_t)0 * Ktot + kc,
                            (unsigned)Ktot, 64u, 32u, 64u);
            __builtin_amdgcn_s_wait_tensorcnt(0);
        }
#else
        for (int e = tid; e < 64 * 32; e += 128) {
            int mm = e >> 5, kk = e & 31;
            As[mm][kk] = wh[(size_t)mm * Ktot + kc + kk];
        }
#endif
        // ---- B tile: bilinear-sampled + modulated, 32x64, 16 elems/thread ----
        #pragma unroll
        for (int r = 0; r < 16; ++r) {
            int e  = tid + r * 128;
            int kk = e >> 6;
            int nn = e & 63;
            int k  = kc + kk;
            int c  = k / KK_, j = k % KK_;
            int i  = (c >> 3) * KK_ + j;    // dg*9 + j
            float py = sPy[i][nn], px = sPx[i][nn];
            float fy = floorf(py), fx = floorf(px);
            int yi = (int)fy, xi = (int)fx;
            float ly = py - fy, lx = px - fx;
            const float* plane = xin + ((size_t)b * NF_ + c) * HW_;
            float v00 = 0.f, v01 = 0.f, v10 = 0.f, v11 = 0.f;
            bool y0ok = (yi >= 0) & (yi < H_);
            bool y1ok = (yi + 1 >= 0) & (yi + 1 < H_);
            bool x0ok = (xi >= 0) & (xi < W_);
            bool x1ok = (xi + 1 >= 0) & (xi + 1 < W_);
            if (y0ok && x0ok) v00 = plane[yi * W_ + xi];
            if (y0ok && x1ok) v01 = plane[yi * W_ + xi + 1];
            if (y1ok && x0ok) v10 = plane[(yi + 1) * W_ + xi];
            if (y1ok && x1ok) v11 = plane[(yi + 1) * W_ + xi + 1];
            float v = (1.f - ly) * ((1.f - lx) * v00 + lx * v01)
                    +        ly  * ((1.f - lx) * v10 + lx * v11);
            Bs[kk][nn] = (_Float16)(v * sMk[i][nn]);
        }
        __syncthreads();

        const int mloc = (lane & 15) + wid * 16;
        const int kb   = (lane < 16) ? 0 : 8;
        v16h afrag;
        {
            v8h lo = *(const v8h*)&As[mloc][kb];
            v8h hi = *(const v8h*)&As[mloc][kb + 16];
            #pragma unroll
            for (int i2 = 0; i2 < 8; ++i2) { afrag[i2] = lo[i2]; afrag[i2 + 8] = hi[i2]; }
        }
        #pragma unroll
        for (int nt = 0; nt < 4; ++nt) {
            const int ncol = nt * 16 + (lane & 15);
            v16h bfrag;
            #pragma unroll
            for (int i2 = 0; i2 < 8; ++i2) { bfrag[i2] = Bs[kb + i2][ncol]; bfrag[i2 + 8] = Bs[kb + 16 + i2][ncol]; }
            acc[nt] = __builtin_amdgcn_wmma_f32_16x16x32_f16(false, afrag, false, bfrag,
                                                             (short)0, acc[nt], false, false);
        }
        __syncthreads();
    }

    const int nn = lane & 15;
    const int mb = wid * 16 + ((lane < 16) ? 0 : 8);
    #pragma unroll
    for (int nt = 0; nt < 4; ++nt) {
        #pragma unroll
        for (int r = 0; r < 8; ++r) {
            int m = mb + r;
            float v = lrelu_f(acc[nt][r] + bias[m]);
            out[((size_t)b * NF_ + m) * HW_ + (size_t)y * W_ + x0 + nt * 16 + nn] = v;
        }
    }
}

// ---------------------------------------------------------------------------
extern "C" void kernel_launch(void* const* d_in, const int* in_sizes, int n_in,
                              void* d_out, int out_size, void* d_ws, size_t ws_size,
                              hipStream_t stream)
{
    (void)in_sizes; (void)n_in; (void)out_size; (void)ws_size;

    const float* nbr   = (const float*)d_in[0];
    const float* ref   = (const float*)d_in[1];
    const float* w1    = (const float*)d_in[2];
    const float* b1    = (const float*)d_in[3];
    const float* w_off = (const float*)d_in[4];
    const float* b_off = (const float*)d_in[5];
    const float* w_dcn = (const float*)d_in[6];
    const float* b_dcn = (const float*)d_in[7];
    const float* wk[6];
    const float* bk[6];
    for (int i = 0; i < 6; ++i) {
        wk[i] = (const float*)d_in[8 + 2 * i];
        bk[i] = (const float*)d_in[9 + 2 * i];
    }

    float* off_a = (float*)d_ws;                        // [B,64,H,W]   16 MB
    float* off_b = off_a + (size_t)NSP * NF_;           // [B,64,H,W]   16 MB
    float* omb   = off_b + (size_t)NSP * NF_;           // [B,216,H,W]  56.6 MB
    _Float16* w1h   = (_Float16*)(omb + (size_t)NSP * (3 * DG_ * KK_));
    _Float16* woffh = w1h + (size_t)64 * 1152;          // 256 x 576 (zero-padded rows)
    _Float16* wdcnh = woffh + (size_t)256 * 576;        // 64 x 576
    float* outp = (float*)d_out;

    const dim3 blk(128);

    // 0) weights -> f16 (pad w_off rows 216 -> 256)
    convert_w_f16<<<(64 * 1152 + 255) / 256, 256, 0, stream>>>(w1, w1h, 64, 64, 1152);
    convert_w_f16<<<(256 * 576 + 255) / 256, 256, 0, stream>>>(w_off, woffh, 216, 256, 576);
    convert_w_f16<<<(64 * 576 + 255) / 256, 256, 0, stream>>>(w_dcn, wdcnh, 64, 64, 576);

    // 1) off = lrelu(conv1(concat(nbr, ref)))          M=64,  K=1152, N=65536
    conv3x3_gemm_wmma<true><<<dim3(NSP / 64, 1), blk, 0, stream>>>(
        nbr, ref, NF_, 2 * NF_, w1h, b1, off_a, NF_);

    // 2) six depthwise 3x3 + lrelu (ping-pong)
    float* cur = off_a;
    float* nxt = off_b;
    for (int i = 0; i < 6; ++i) {
        dwconv3x3_lrelu<<<(B_ * NF_ * HW_ + 255) / 256, 256, 0, stream>>>(cur, wk[i], bk[i], nxt);
        float* t = cur; cur = nxt; nxt = t;
    }

    // 3) om = conv_off(off)                             M=216 (pad 256), K=576
    conv3x3_gemm_wmma<false><<<dim3(NSP / 64, 4), blk, 0, stream>>>(
        cur, cur, NF_, NF_, woffh, b_off, omb, 3 * DG_ * KK_);

    // 4) out = lrelu(deform_conv(nbr, om, w_dcn))       M=64,  K=576
    deform_conv_wmma<<<dim3(NSP / 64), blk, 0, stream>>>(nbr, omb, wdcnh, b_dcn, outp);
}
